// Varkeys_25778393710918
// MI455X (gfx1250) — matrix operand order repaired and checked by reading, not compiled, not run on hardware
//
#include <hip/hip_runtime.h>

// ---------------------------------------------------------------------------
// MI455X (gfx1250, wave32) fused VarKeys kernel, v3.
//   GEMM1 (keys @ x^T) -> RBF reciprocal -> GEMM2 (o^T @ values), fused so the
//   204M-element 'o' matrix never touches HBM. bf16 WMMA with f32 accumulate;
//   exact-f32 norm terms keep the distance computation accurate.
//   v3: replace the IEEE-exact division chains in the reciprocal kernel with
//   v_fma + v_rcp_f32 (hardware approx, ~1 ulp << bf16 quantization), cutting
//   ~25 VALU ops/element to 2 on the 204M-element hot path.
// ---------------------------------------------------------------------------

typedef __attribute__((ext_vector_type(16))) __bf16 v16bf;
typedef __attribute__((ext_vector_type(8)))  __bf16 v8bf;
typedef __attribute__((ext_vector_type(8)))  float  v8f;

#define B_TOT   4096      // batch
#define KS      128       // keysize
#define D_TOT   50000     // dict size
#define C_TOT   1000      // categories
#define BM      32        // batch rows per workgroup (2 x 16 M-tiles)
#define MT      2         // 16-row M-tiles
#define DK      64        // dict chunk per iteration
#define CB      512       // category columns per workgroup (grid.y = 2)
#define NT      4         // 16-wide category tiles per wave (CB / 8 waves / 16)
#define KSTR    (KS + 8)  // padded LDS row stride (halves)
#define DKSTR   (DK + 8)  // padded LDS row stride (halves)

// A-operand (16 x 32 bf16) tile load from row-major [M][K] memory.
// ISA 7.12.2: lanes 0-15 hold K=0..7 & 16..23, lanes 16-31 hold K=8..15 & 24..31.
__device__ __forceinline__ v16bf load_a_tile(const __bf16* base, int stride, int lane) {
  const __bf16* p = base + (lane & 15) * stride + ((lane >> 4) << 3);
  v8bf lo = *(const v8bf*)(p);
  v8bf hi = *(const v8bf*)(p + 16);
  return __builtin_shufflevector(lo, hi, 0,1,2,3,4,5,6,7,8,9,10,11,12,13,14,15);
}

// B-operand (32 x 16 bf16) tile load from row-major [N][K] (i.e. B^T) memory.
// Layout: lanes 0-15 hold K=0..7 & 16..23, lanes 16-31 hold K=8..15 & 24..31.
__device__ __forceinline__ v16bf load_b_tile(const __bf16* base, int stride, int lane) {
  const __bf16* p = base + (lane & 15) * stride + ((lane >> 4) << 3);
  v8bf lo = *(const v8bf*)(p);
  v8bf hi = *(const v8bf*)(p + 16);
  return __builtin_shufflevector(lo, hi, 0,1,2,3,4,5,6,7,8,9,10,11,12,13,14,15);
}

__global__ __launch_bounds__(256)
void varkeys_fused_kernel(const float* __restrict__ x,
                          const float* __restrict__ keys,
                          const float* __restrict__ values,
                          float* __restrict__ out) {
  __shared__ __align__(16) __bf16 lds_x[BM * KSTR];       //  8.7 KB  x tile (bf16, row-major)
  __shared__ __align__(16) __bf16 lds_keys[DK * KSTR];    // 17.0 KB  keys chunk (bf16)
  __shared__ __align__(16) __bf16 lds_valsT[CB * DKSTR];  // 72.0 KB  values chunk, transposed [cat][dict]
  __shared__ __align__(16) __bf16 lds_oT[BM * DKSTR];     //  4.6 KB  o^T tile [batch][dict]
  __shared__ float lds_kk[DK];                            // ||k||^2 per chunk row (exact f32)
  __shared__ float lds_xx[BM];                            // ||x||^2 per batch row (exact f32)

  const int tid  = threadIdx.x;
  const int lane = tid & 31;
  const int wave = tid >> 5;
  const int b0   = blockIdx.x * BM;
  const int c0   = blockIdx.y * CB;

  // ---- stage x tile as bf16 + exact f32 row norms --------------------------
#pragma unroll
  for (int jj = 0; jj < 2; ++jj) {
    const int row = (tid >> 4) + jj * 16;
    const int col = (tid & 15) << 3;
    const float* src = x + (size_t)(b0 + row) * KS + col;
    float4 a = *(const float4*)(src);
    float4 b = *(const float4*)(src + 4);
    __bf16* dst = &lds_x[row * KSTR + col];
    dst[0] = (__bf16)a.x; dst[1] = (__bf16)a.y; dst[2] = (__bf16)a.z; dst[3] = (__bf16)a.w;
    dst[4] = (__bf16)b.x; dst[5] = (__bf16)b.y; dst[6] = (__bf16)b.z; dst[7] = (__bf16)b.w;
  }
  if (tid < BM) {
    const float* src = x + (size_t)(b0 + tid) * KS;
    float ssq = 0.0f;
    for (int k = 0; k < KS; ++k) { float v = src[k]; ssq += v * v; }
    lds_xx[tid] = ssq;
  }
  __syncthreads();

  // per-wave GEMM1 tile assignment: dict tile (wave&3), batch tile (wave>>2)
  const int mw = (wave & 3) * 16;   // dict rows within chunk
  const int nw = (wave >> 2) * 16;  // batch rows within tile

  // hoist GEMM1 B operand (x^T rows nw..nw+15) into registers
  v16bf xb[4];
#pragma unroll
  for (int kc = 0; kc < 4; ++kc)
    xb[kc] = load_b_tile(&lds_x[nw * KSTR + kc * 32], KSTR, lane);

  v8f acc[MT][NT] = {};  // 2 x 4 KV accumulator tiles, live across the dict loop

  for (int d0 = 0; d0 < D_TOT; d0 += DK) {
    __syncthreads();  // previous iteration's GEMM2 reads of valsT/oT are done

    // hint next chunk into cache (global_prefetch_b8)
    if ((tid & 63) == 0) {
      int dn = d0 + DK;
      if (dn < D_TOT) {
        __builtin_prefetch(keys + (size_t)dn * KS, 0, 1);
        __builtin_prefetch(values + (size_t)dn * C_TOT + c0, 0, 1);
      }
    }

    // ---- stage keys chunk (bf16) + per-row ||k||^2 via wave32 reduction ----
#pragma unroll
    for (int j = 0; j < 8; ++j) {
      const int q  = tid + j * 256;   // quad id; one wave covers one key row
      const int r  = q >> 5;
      const int cq = (q & 31) << 2;
      const int dg = d0 + r;
      float4 a = make_float4(0.0f, 0.0f, 0.0f, 0.0f);
      if (dg < D_TOT) a = *(const float4*)(keys + (size_t)dg * KS + cq);
      __bf16* dst = &lds_keys[r * KSTR + cq];
      dst[0] = (__bf16)a.x; dst[1] = (__bf16)a.y; dst[2] = (__bf16)a.z; dst[3] = (__bf16)a.w;
      float s = a.x * a.x + a.y * a.y + a.z * a.z + a.w * a.w;
      s += __shfl_xor(s, 16); s += __shfl_xor(s, 8); s += __shfl_xor(s, 4);
      s += __shfl_xor(s, 2);  s += __shfl_xor(s, 1);
      if (lane == 0) lds_kk[r] = s;
    }

    // ---- stage values chunk transposed: lds_valsT[cat][dict] (bf16) --------
    for (int j = 0; j < 32; ++j) {
      const int q  = tid + j * 256;
      const int r  = q >> 7;
      const int cq = (q & 127) << 2;
      const int dg = d0 + r;
      const int cg = c0 + cq;
      float4 a = make_float4(0.0f, 0.0f, 0.0f, 0.0f);
      if (dg < D_TOT) {
        const float* src = values + (size_t)dg * C_TOT + cg;
        if (cg + 3 < C_TOT) {
          a = *(const float4*)(src);
        } else {
          if (cg + 0 < C_TOT) a.x = src[0];
          if (cg + 1 < C_TOT) a.y = src[1];
          if (cg + 2 < C_TOT) a.z = src[2];
          if (cg + 3 < C_TOT) a.w = src[3];
        }
      }
      lds_valsT[(cq + 0) * DKSTR + r] = (__bf16)a.x;
      lds_valsT[(cq + 1) * DKSTR + r] = (__bf16)a.y;
      lds_valsT[(cq + 2) * DKSTR + r] = (__bf16)a.z;
      lds_valsT[(cq + 3) * DKSTR + r] = (__bf16)a.w;
    }
    __syncthreads();

    // ---- GEMM1: each wave computes one 16x16 S tile, then the RBF kernel ---
    {
      v8f s = {};
#pragma unroll
      for (int kc = 0; kc < 4; ++kc) {
        v16bf A = load_a_tile(&lds_keys[mw * KSTR + kc * 32], KSTR, lane);
        s = __builtin_amdgcn_wmma_f32_16x16x32_bf16(false, A, false, xb[kc],
                                                    (short)0, s, false, false);
      }
      const int hh = lane >> 4;   // C/D layout: M = v + 8*(lane/16), N = lane%16
      const int n  = lane & 15;
      const float xxn = lds_xx[nw + n];
      v8bf ob;
#pragma unroll
      for (int v = 0; v < 8; ++v) {
        const int m = mw + hh * 8 + v;
        float d = lds_kk[m] - 2.0f * s[v] + xxn;
        // o = 1/(d/1e4 + 1e-4): v_fma + v_rcp_f32 (approx rcp is ~1 ulp,
        // far below the bf16 quantization applied immediately after).
        float den = fmaf(d, 1.0e-4f, 1.0e-4f);
        float o = __builtin_amdgcn_rcpf(den);
        if (d0 + m >= D_TOT) o = 0.0f;  // zero-pad dict tail
        ob[v] = (__bf16)o;
      }
      // contiguous 16B store lands o^T directly in GEMM2 A-operand layout
      *(v8bf*)(&lds_oT[(nw + n) * DKSTR + mw + hh * 8]) = ob;
    }
    __syncthreads();

    // ---- GEMM2: KV(32 x CB) += o^T(32 x 64) @ values(64 x CB) --------------
    {
      v16bf A_[MT][2];
#pragma unroll
      for (int mt = 0; mt < MT; ++mt)
#pragma unroll
        for (int kc = 0; kc < 2; ++kc)
          A_[mt][kc] = load_a_tile(&lds_oT[(mt * 16) * DKSTR + kc * 32], DKSTR, lane);

      const __bf16* bb = &lds_valsT[(wave * 64) * DKSTR];
      v16bf B0 = load_b_tile(bb,      DKSTR, lane);
      v16bf B1 = load_b_tile(bb + 32, DKSTR, lane);
#pragma unroll
      for (int t = 0; t < NT; ++t) {
        v16bf C0 = B0, C1 = B1;
        if (t + 1 < NT) {  // pipeline: issue next tile's loads before the WMMAs
          const __bf16* nb = bb + (size_t)(t + 1) * 16 * DKSTR;
          B0 = load_b_tile(nb,      DKSTR, lane);
          B1 = load_b_tile(nb + 32, DKSTR, lane);
        }
        acc[0][t] = __builtin_amdgcn_wmma_f32_16x16x32_bf16(false, A_[0][0], false, C0,
                                                            (short)0, acc[0][t], false, false);
        acc[1][t] = __builtin_amdgcn_wmma_f32_16x16x32_bf16(false, A_[1][0], false, C0,
                                                            (short)0, acc[1][t], false, false);
        acc[0][t] = __builtin_amdgcn_wmma_f32_16x16x32_bf16(false, A_[0][1], false, C1,
                                                            (short)0, acc[0][t], false, false);
        acc[1][t] = __builtin_amdgcn_wmma_f32_16x16x32_bf16(false, A_[1][1], false, C1,
                                                            (short)0, acc[1][t], false, false);
      }
    }
  }

  // ---- write unnormalized KV tiles (rownorm kernel finishes the job) -------
  {
    const int hh = lane >> 4;
    const int n  = lane & 15;
#pragma unroll
    for (int mt = 0; mt < MT; ++mt) {
#pragma unroll
      for (int t = 0; t < NT; ++t) {
        const int cg = c0 + wave * 64 + t * 16 + n;
        if (cg < C_TOT) {
#pragma unroll
          for (int v = 0; v < 8; ++v)
            out[(size_t)(b0 + mt * 16 + hh * 8 + v) * C_TOT + cg] = acc[mt][t][v];
        }
      }
    }
  }
}

// In-place row normalization: one workgroup per batch row (32 MB traffic).
__global__ __launch_bounds__(256)
void rownorm_kernel(float* __restrict__ out) {
  const int row = blockIdx.x;
  const int tid = threadIdx.x;
  float* p = out + (size_t)row * C_TOT;

  float s = 0.0f;
  for (int c = tid; c < C_TOT; c += 256) s += p[c];
  s += __shfl_xor(s, 16); s += __shfl_xor(s, 8); s += __shfl_xor(s, 4);
  s += __shfl_xor(s, 2);  s += __shfl_xor(s, 1);

  __shared__ float red[8];
  __shared__ float invs;
  if ((tid & 31) == 0) red[tid >> 5] = s;
  __syncthreads();
  if (tid == 0) {
    float t = 0.0f;
    for (int i = 0; i < 8; ++i) t += red[i];
    invs = 1.0f / t;  // once per row: keep IEEE-exact
  }
  __syncthreads();
  const float inv = invs;
  for (int c = tid; c < C_TOT; c += 256) p[c] *= inv;
}

extern "C" void kernel_launch(void* const* d_in, const int* in_sizes, int n_in,
                              void* d_out, int out_size, void* d_ws, size_t ws_size,
                              hipStream_t stream) {
  (void)in_sizes; (void)n_in; (void)out_size; (void)d_ws; (void)ws_size;
  const float* x      = (const float*)d_in[0];
  const float* keys   = (const float*)d_in[1];
  const float* values = (const float*)d_in[2];
  float* out = (float*)d_out;

  dim3 grid(B_TOT / BM, (C_TOT + CB - 1) / CB);  // (128, 2)
  varkeys_fused_kernel<<<grid, dim3(256), 0, stream>>>(x, keys, values, out);
  rownorm_kernel<<<dim3(B_TOT), dim3(256), 0, stream>>>(out);
}
